// CMPModel_13065290514577
// MI455X (gfx1250) — matrix operand order repaired and checked by reading, not compiled, not run on hardware
//
#include <hip/hip_runtime.h>
#include <hip/hip_bf16.h>

// ---------------------------------------------------------------------------
// CMP density matrix:  real = Rw^T R + Iw^T I ;  imag = Iw^T R - Rw^T I
// (Rw = w*word_emb[q], Iw = w*pos*cmp_emb[q], R = word_emb[q], I = pos*cmp_emb[q])
// B=64 batched 256x256x128 GEMMs on v_wmma_f32_16x16x32_f16 (f32 accumulate).
// Memory-bound target: ~50MB HBM traffic @ 23.3 TB/s; f16 WMMA keeps compute
// (~4.3 GFLOP) well under the memory floor, unlike the fp32 16x16x4 path.
// ---------------------------------------------------------------------------

typedef __attribute__((ext_vector_type(16))) _Float16 v16h;
typedef __attribute__((ext_vector_type(8)))  _Float16 v8h;
typedef __attribute__((ext_vector_type(8)))  float    v8f;

#define S_LEN  128          // sequence length (= GEMM K)
#define D_FULL 256          // embedding dim
#define TILE   64           // output tile per block (64x64)
#define LSTR   136          // LDS row stride in halves (128 + 8 pad: keeps 16B
                            // alignment, staggers banks)
#define TSZ    (TILE * LSTR)

// A fragment (16x32 f16, M x K): lane<16 holds row M=lane, K = {0..7,16..23};
// lane>=16 holds row M=lane-16, K = {8..15,24..31}.  Tile is feature-major
// [feat][s], so each half-fragment is 8 contiguous halves -> ds_load_b128.
__device__ __forceinline__ v16h frag_a(const _Float16* t, int featBase, int k0, int lane) {
  const _Float16* p = t + (featBase + (lane & 15)) * LSTR + k0 + ((lane & 16) ? 8 : 0);
  v8h lo = *(const v8h*)p;
  v8h hi = *(const v8h*)(p + 16);
  return __builtin_shufflevector(lo, hi, 0,1,2,3,4,5,6,7,8,9,10,11,12,13,14,15);
}

// B fragment (32x16 f16, K x N): lane<16 holds column N=lane, K = {0..7,16..23};
// lane>=16 holds column N=lane-16, K = {8..15,24..31}.
__device__ __forceinline__ v16h frag_b(const _Float16* t, int featBase, int k0, int lane) {
  const _Float16* p = t + (featBase + (lane & 15)) * LSTR + k0 + ((lane & 16) ? 8 : 0);
  v8h lo = *(const v8h*)p;
  v8h hi = *(const v8h*)(p + 16);
  return __builtin_shufflevector(lo, hi, 0,1,2,3,4,5,6,7,8,9,10,11,12,13,14,15);
}

__device__ __forceinline__ v8f wmma_acc(v16h a, v16h b, v8f c) {
  // (neg_a, A, neg_b, B, c_mod, C, reuse_a, reuse_b)
  return __builtin_amdgcn_wmma_f32_16x16x32_f16(false, a, false, b, (short)0, c, false, false);
}

__global__ __launch_bounds__(256)
void cmp_density_kernel(const int* __restrict__ questions,      // [B,S]
                        const float* __restrict__ q_position,   // [B,S]
                        const float* __restrict__ word_emb,     // [V,D]
                        const float* __restrict__ cmp_emb,      // [V,D]
                        const float* __restrict__ weighted_q,   // [S]
                        float* __restrict__ out)                // [2,B,D,D]
{
  extern __shared__ _Float16 lds[];
  _Float16* sRw = lds;            // A-side: w * word      [64 d][136 s]
  _Float16* sIw = lds + 1 * TSZ;  // A-side: w * pos * cmp
  _Float16* sR  = lds + 2 * TSZ;  // B-side: word          [64 e][136 s]
  _Float16* sI  = lds + 3 * TSZ;  // B-side: pos * cmp

  const int et = blockIdx.x;      // e tile (0..3)
  const int dt = blockIdx.y;      // d tile (0..3)
  const int b  = blockIdx.z;      // batch  (0..63)
  const int d0 = dt * TILE;
  const int e0 = et * TILE;
  const int tid = threadIdx.x;

  // -------- Phase 1: gather + weight-fold + f32->f16, transposed into LDS ----
  // 2048 work items = 128 s-rows x 16 float4 feature chunks; uniform 8 iters.
  for (int i = tid; i < S_LEN * 16; i += 256) {
    const int s = i >> 4;
    const int c = (i & 15) << 2;                 // feature offset 0..60 step 4
    const int   q   = questions[b * S_LEN + s];
    const float pos = q_position[b * S_LEN + s];
    const float w   = weighted_q[s];
    const float pw  = w * pos;

    const float4 wa = *(const float4*)(word_emb + (size_t)q * D_FULL + d0 + c);
    const float4 ca = *(const float4*)(cmp_emb  + (size_t)q * D_FULL + d0 + c);
    const float4 wb = *(const float4*)(word_emb + (size_t)q * D_FULL + e0 + c);
    const float4 cb = *(const float4*)(cmp_emb  + (size_t)q * D_FULL + e0 + c);

    const float* wap = (const float*)&wa;
    const float* cap = (const float*)&ca;
    const float* wbp = (const float*)&wb;
    const float* cbp = (const float*)&cb;
#pragma unroll
    for (int j = 0; j < 4; ++j) {
      sRw[(c + j) * LSTR + s] = (_Float16)(w   * wap[j]);
      sIw[(c + j) * LSTR + s] = (_Float16)(pw  * cap[j]);
      sR [(c + j) * LSTR + s] = (_Float16)(       wbp[j]);
      sI [(c + j) * LSTR + s] = (_Float16)(pos * cbp[j]);
    }
  }
  __syncthreads();

  // -------- Phase 2: 8 waves; wave owns 1 m-tile x 2 n-tiles, K=128 ---------
  const int lane = tid & 31;
  const int wv   = tid >> 5;               // 0..7
  const int mt   = (wv & 3) * 16;          // A feature base (d)
  const int nb   = (wv >> 2) * 32;         // B feature base (e): tiles nb, nb+16

  v8f accR0 = {}, accP0 = {}, accM0 = {};  // real, Iw^T R, Rw^T I (tile 0)
  v8f accR1 = {}, accP1 = {}, accM1 = {};  // tile 1

#pragma unroll
  for (int k0 = 0; k0 < S_LEN; k0 += 32) {
    const v16h aR  = frag_a(sRw, mt, k0, lane);
    const v16h aI  = frag_a(sIw, mt, k0, lane);
    const v16h bR0 = frag_b(sR,  nb,      k0, lane);
    const v16h bI0 = frag_b(sI,  nb,      k0, lane);
    const v16h bR1 = frag_b(sR,  nb + 16, k0, lane);
    const v16h bI1 = frag_b(sI,  nb + 16, k0, lane);

    accR0 = wmma_acc(aR, bR0, accR0);
    accR0 = wmma_acc(aI, bI0, accR0);
    accP0 = wmma_acc(aI, bR0, accP0);
    accM0 = wmma_acc(aR, bI0, accM0);

    accR1 = wmma_acc(aR, bR1, accR1);
    accR1 = wmma_acc(aI, bI1, accR1);
    accP1 = wmma_acc(aI, bR1, accP1);
    accM1 = wmma_acc(aR, bI1, accM1);
  }

  const v8f im0 = accP0 - accM0;           // imag = M - M^T done per-element
  const v8f im1 = accP1 - accM1;

  // -------- Epilogue: C/D layout — VGPR r: lane<16 -> (M=r,N=lane),
  //                                 lane>=16 -> (M=r+8,N=lane-16) -------------
  const int Moff = (lane & 16) ? 8 : 0;
  const int N    = lane & 15;
  float* realBase = out + (((size_t)b * D_FULL) + d0 + mt) * D_FULL + e0;
  float* imagBase = realBase + (size_t)64 * D_FULL * D_FULL;  // B*D*D offset

#pragma unroll
  for (int r = 0; r < 8; ++r) {
    const int rowo = (r + Moff) * D_FULL;
    realBase[rowo + nb +      N] = accR0[r];
    realBase[rowo + nb + 16 + N] = accR1[r];
    imagBase[rowo + nb +      N] = im0[r];
    imagBase[rowo + nb + 16 + N] = im1[r];
  }
}

extern "C" void kernel_launch(void* const* d_in, const int* in_sizes, int n_in,
                              void* d_out, int out_size, void* d_ws, size_t ws_size,
                              hipStream_t stream) {
  (void)in_sizes; (void)n_in; (void)out_size; (void)d_ws; (void)ws_size;
  const int*   questions  = (const int*)  d_in[0];
  const float* q_position = (const float*)d_in[1];
  const float* word_emb   = (const float*)d_in[2];
  const float* cmp_emb    = (const float*)d_in[3];
  const float* weighted_q = (const float*)d_in[4];
  float* out = (float*)d_out;

  const size_t shmem = (size_t)4 * TSZ * sizeof(_Float16);   // 69632 B < 320KB/WG
  // Unconditional, idempotent, host-side, deterministic (no static guards):
  hipFuncSetAttribute((const void*)cmp_density_kernel,
                      hipFuncAttributeMaxDynamicSharedMemorySize, (int)shmem);

  dim3 grid(D_FULL / TILE, D_FULL / TILE, 64);  // 4 x 4 x 64 = 1024 blocks
  cmp_density_kernel<<<grid, dim3(256), shmem, stream>>>(
      questions, q_position, word_emb, cmp_emb, weighted_q, out);
}
